// GAT_pool_56581899157772
// MI455X (gfx1250) — matrix operand order repaired and checked by reading, not compiled, not run on hardware
//
#include <hip/hip_runtime.h>
#include <hip/hip_bf16.h>

typedef _Float16 h8   __attribute__((ext_vector_type(8)));
typedef _Float16 v16h __attribute__((ext_vector_type(16)));
typedef float    v8f  __attribute__((ext_vector_type(8)));

#define WAVES_PER_BLOCK 8
#define NT 4            // 16-wide column tiles per wave -> 16x64 output per wave

// ---------------- utility kernels ----------------

__global__ void fill_kernel(float* __restrict__ p, float v, int n) {
    int i = blockIdx.x * blockDim.x + threadIdx.x;
    if (i < n) p[i] = v;
}

__global__ void cast_f16_kernel(const float* __restrict__ s, _Float16* __restrict__ d, int n) {
    int i = blockIdx.x * blockDim.x + threadIdx.x;
    if (i < n) d[i] = (_Float16)s[i];
}

// W [R x Cc] row-major (f32)  ->  Wt [Cc x R] row-major (f16)
__global__ void castT_f16_kernel(const float* __restrict__ W, _Float16* __restrict__ Wt,
                                 int R, int Cc) {
    int i = blockIdx.x * blockDim.x + threadIdx.x;
    if (i < R * Cc) {
        int r = i / Cc, c = i % Cc;
        Wt[(size_t)c * R + r] = (_Float16)W[i];
    }
}

// ---------------- WMMA GEMM: C[MxNc] f32 = A[MxK] f16 * B[KxNc] f16 (B given transposed) ----
// One wave computes a 16x64 tile; A fragment reused across 4 WMMAs per K-step.
// Fragment layouts per CDNA5 ISA 7.12.2 (16-bit A 16x32, B 32x16, f32 C 16x16).
__global__ __launch_bounds__(256) void wmma_gemm_f16_kernel(
        const _Float16* __restrict__ A,   // M x K
        const _Float16* __restrict__ Bt,  // Nc x K   (row n holds column n of B)
        float* __restrict__ C,            // M x Nc
        int M, int K, int Nc) {
    int wave = threadIdx.x >> 5;
    int lane = threadIdx.x & 31;
    int rowTile = blockIdx.x * WAVES_PER_BLOCK + wave;
    if (rowTile * 16 >= M) return;                 // wave-uniform exit, no barriers used
    int colBase = blockIdx.y * (NT * 16);

    int m  = lane & 15;
    int hi = lane >> 4;

    const _Float16* arow  = A  + (size_t)(rowTile * 16 + m) * K;
    const _Float16* bbase = Bt + (size_t)(colBase + m) * K;

    union VF { v8f v; float f[8]; } acc[NT];
#pragma unroll
    for (int t = 0; t < NT; ++t)
        acc[t].v = (v8f){0.f,0.f,0.f,0.f,0.f,0.f,0.f,0.f};

    for (int k0 = 0; k0 < K; k0 += 32) {
        union AF { v16h v; h8 h[2]; } af;
        // A lane holds row m; halves 0..7 -> K = k0+hi*8+j, halves 8..15 -> K = k0+16+hi*8+j
        af.h[0] = *(const h8*)(arow + k0 + hi * 8);
        af.h[1] = *(const h8*)(arow + k0 + 16 + hi * 8);
#pragma unroll
        for (int t = 0; t < NT; ++t) {
            union AF bf;
            // B lane holds column (colBase+t*16+m); halves 0..7 -> K = k0+hi*8+j, halves 8..15 -> K = k0+16+hi*8+j
            const _Float16* bp = bbase + (size_t)t * 16 * K + k0 + hi * 8;
            bf.h[0] = *(const h8*)(bp);
            bf.h[1] = *(const h8*)(bp + 16);
            acc[t].v = __builtin_amdgcn_wmma_f32_16x16x32_f16(
                false, af.v, false, bf.v, (short)0, acc[t].v, false, false);
        }
    }

#pragma unroll
    for (int t = 0; t < NT; ++t) {
        int col = colBase + t * 16 + m;
#pragma unroll
        for (int v = 0; v < 8; ++v) {
            int row = rowTile * 16 + v + hi * 8;   // C layout: VGPR v -> M=v (+8 for hi lanes)
            C[(size_t)row * Nc + col] = acc[t].f[v];
        }
    }
}

// ---------------- attention kernels ----------------

// one wave per (node, head): alpha_s/alpha_d dot products over C channels
__global__ void alpha_kernel(const float* __restrict__ h,
                             const float* __restrict__ a_src, const float* __restrict__ a_dst,
                             float* __restrict__ als, float* __restrict__ ald,
                             int N, int H, int C) {
    int lane = threadIdx.x & 31;
    int gw = blockIdx.x * (blockDim.x >> 5) + (threadIdx.x >> 5);
    if (gw >= N * H) return;
    int n = gw / H, hd = gw - n * H;
    const float* hp  = h + (size_t)n * H * C + (size_t)hd * C;
    const float* asp = a_src + (size_t)hd * C;
    const float* adp = a_dst + (size_t)hd * C;
    float ss = 0.f, sd = 0.f;
    for (int c = lane; c < C; c += 32) {
        float v = hp[c];
        ss += v * asp[c];
        sd += v * adp[c];
    }
    for (int off = 16; off > 0; off >>= 1) {
        ss += __shfl_xor(ss, off, 32);
        sd += __shfl_xor(sd, off, 32);
    }
    if (lane == 0) { als[gw] = ss; ald[gw] = sd; }
}

__device__ inline void atomicMaxF(float* addr, float val) {
    unsigned* ua = (unsigned*)addr;
    unsigned old = *ua;
    while (__uint_as_float(old) < val) {
        unsigned assumed = old;
        old = atomicCAS(ua, assumed, __float_as_uint(val));
        if (old == assumed) break;
    }
}

// leaky-relu logit per (edge, head) + segment max over dst
__global__ void edge_logit_max_kernel(const int* __restrict__ ei, int E, int N, int H,
                                      const float* __restrict__ als, const float* __restrict__ ald,
                                      float* __restrict__ elog, float* __restrict__ mx) {
    int idx = blockIdx.x * blockDim.x + threadIdx.x;
    int EE = E + N;
    if (idx >= EE * H) return;
    int e = idx / H, hd = idx - e * H;
    int s = (e < E) ? ei[e]     : (e - E);
    int d = (e < E) ? ei[E + e] : (e - E);
    float v = als[s * H + hd] + ald[d * H + hd];
    v = (v > 0.f) ? v : 0.2f * v;
    elog[idx] = v;
    atomicMaxF(&mx[d * H + hd], v);
}

// exp(e - m[dst]) + segment sum
__global__ void edge_exp_sum_kernel(const int* __restrict__ ei, int E, int N, int H,
                                    const float* __restrict__ mx,
                                    float* __restrict__ elog, float* __restrict__ dn) {
    int idx = blockIdx.x * blockDim.x + threadIdx.x;
    int EE = E + N;
    if (idx >= EE * H) return;
    int e = idx / H, hd = idx - e * H;
    int d = (e < E) ? ei[E + e] : (e - E);
    float ex = __expf(elog[idx] - mx[d * H + hd]);
    elog[idx] = ex;
    atomicAdd(&dn[d * H + hd], ex);
}

// one wave per (edge, head): out[dst] += alpha * h[src]  (float4 gather + f32 atomics)
__global__ void aggregate_kernel(const int* __restrict__ ei, int E, int N, int H, int C,
                                 const float* __restrict__ h, const float* __restrict__ elog,
                                 const float* __restrict__ dn, float* __restrict__ out) {
    int lane = threadIdx.x & 31;
    int gw = blockIdx.x * (blockDim.x >> 5) + (threadIdx.x >> 5);
    int EE = E + N;
    if (gw >= EE * H) return;
    int e = gw / H, hd = gw - e * H;
    int s = (e < E) ? ei[e]     : (e - E);
    int d = (e < E) ? ei[E + e] : (e - E);
    float alpha = elog[gw] / (dn[d * H + hd] + 1e-16f);
    const float4* hp = (const float4*)(h + (size_t)s * H * C + (size_t)hd * C);
    float* op = out + (size_t)d * H * C + (size_t)hd * C;
    int nv = C >> 2;
    for (int i = lane; i < nv; i += 32) {
        float4 hv = hp[i];
        atomicAdd(op + 4 * i + 0, alpha * hv.x);
        atomicAdd(op + 4 * i + 1, alpha * hv.y);
        atomicAdd(op + 4 * i + 2, alpha * hv.z);
        atomicAdd(op + 4 * i + 3, alpha * hv.w);
    }
}

// relu(in + b) and cast to f16 (input to layer-2 GEMM)
__global__ void bias_relu_cast_kernel(const float* __restrict__ in, const float* __restrict__ b,
                                      _Float16* __restrict__ out, int N, int F) {
    int i = blockIdx.x * blockDim.x + threadIdx.x;
    if (i < N * F) {
        int f = i % F;
        float v = in[i] + b[f];
        out[i] = (_Float16)(v > 0.f ? v : 0.f);
    }
}

// per-graph sum of (out2 + b2) and node counts
__global__ void pool_accum_kernel(const float* __restrict__ o2, const float* __restrict__ b2,
                                  const int* __restrict__ batch,
                                  float* __restrict__ psum, float* __restrict__ pcnt,
                                  int N, int C) {
    int i = blockIdx.x * blockDim.x + threadIdx.x;
    if (i < N * C) {
        int n = i / C, c = i - n * C;
        int g = batch[n];
        atomicAdd(&psum[(size_t)g * C + c], o2[i] + b2[c]);
        if (c == 0) atomicAdd(&pcnt[g], 1.0f);
    }
}

// mean pool + fc1(relu) + fc2 : one block per graph, blockDim = hidden (64)
__global__ void final_mlp_kernel(const float* __restrict__ psum, const float* __restrict__ pcnt,
                                 const float* __restrict__ fc1W, const float* __restrict__ fc1b,
                                 const float* __restrict__ fc2W, const float* __restrict__ fc2b,
                                 float* __restrict__ out, int C, int Hd) {
    __shared__ float sh[256];
    __shared__ float red[2];
    int g = blockIdx.x, t = threadIdx.x;
    float cnt = pcnt[g];
    if (cnt < 1.f) cnt = 1.f;
    for (int c = t; c < C; c += blockDim.x) sh[c] = psum[(size_t)g * C + c] / cnt;
    if (t < 2) red[t] = 0.f;
    __syncthreads();
    float acc = 0.f;
    for (int c = 0; c < C; ++c) acc += sh[c] * fc1W[c * Hd + t];
    acc += fc1b[t];
    acc = acc > 0.f ? acc : 0.f;
    float p = acc * fc2W[t];
    for (int off = 16; off > 0; off >>= 1) p += __shfl_xor(p, off, 32);
    if ((t & 31) == 0) red[t >> 5] = p;
    __syncthreads();
    if (t == 0) out[g] = red[0] + red[1] + fc2b[0];
}

// ---------------- host-side launch ----------------

extern "C" void kernel_launch(void* const* d_in, const int* in_sizes, int n_in,
                              void* d_out, int out_size, void* d_ws, size_t ws_size,
                              hipStream_t stream) {
    const float* x      = (const float*)d_in[0];
    const int*   ei     = (const int*)  d_in[1];
    const int*   batch  = (const int*)  d_in[2];
    const float* W1     = (const float*)d_in[3];
    const float* a_src1 = (const float*)d_in[4];
    const float* a_dst1 = (const float*)d_in[5];
    const float* b1     = (const float*)d_in[6];
    const float* W2     = (const float*)d_in[7];
    const float* a_src2 = (const float*)d_in[8];
    const float* a_dst2 = (const float*)d_in[9];
    const float* b2     = (const float*)d_in[10];
    const float* fc1W   = (const float*)d_in[11];
    const float* fc1b   = (const float*)d_in[12];
    const float* fc2W   = (const float*)d_in[13];
    const float* fc2b   = (const float*)d_in[14];

    const int N  = in_sizes[2];          // 20000
    const int D  = in_sizes[0] / N;      // 256
    const int E  = in_sizes[1] / 2;      // 160000
    const int HC = in_sizes[6];          // 512 (= H*C)
    const int C  = in_sizes[10];         // 128
    const int H  = HC / C;               // 4
    const int Hd = in_sizes[12];         // 64
    const int G  = out_size;             // 64
    const int EE = E + N;                // edges + self loops

    char* ws = (char*)d_ws;
    size_t off = 0;
    auto alloc = [&](size_t bytes) -> void* {
        void* p = ws + off;
        off = (off + bytes + 255) & ~(size_t)255;
        return p;
    };

    _Float16* x_h  = (_Float16*)alloc((size_t)N * D * 2);
    _Float16* Wt1  = (_Float16*)alloc((size_t)HC * D * 2);
    _Float16* Wt2  = (_Float16*)alloc((size_t)C * HC * 2);
    float* as1 = (float*)alloc((size_t)N * H * 4);
    float* ad1 = (float*)alloc((size_t)N * H * 4);
    float* m1  = (float*)alloc((size_t)N * H * 4);
    float* dn1 = (float*)alloc((size_t)N * H * 4);
    float* el1 = (float*)alloc((size_t)EE * H * 4);
    float* as2 = (float*)alloc((size_t)N * 4);
    float* ad2 = (float*)alloc((size_t)N * 4);
    float* m2  = (float*)alloc((size_t)N * 4);
    float* dn2 = (float*)alloc((size_t)N * 4);
    float* el2 = (float*)alloc((size_t)EE * 4);
    float* psum = (float*)alloc((size_t)G * C * 4);
    float* pcnt = (float*)alloc((size_t)G * 4);
    float* h1   = (float*)alloc((size_t)N * HC * 4);
    float* out1 = (float*)alloc((size_t)N * HC * 4);
    _Float16* h1r = (_Float16*)alloc((size_t)N * HC * 2);
    float* h2   = (float*)alloc((size_t)N * C * 4);
    float* out2 = (float*)alloc((size_t)N * C * 4);
    (void)ws_size; (void)n_in;

    const int TPB = 256;
    auto blk = [&](long n) { return (int)((n + TPB - 1) / TPB); };

    // precision conversion (weights transposed so B fragments load contiguously)
    cast_f16_kernel<<<blk((long)N * D), TPB, 0, stream>>>(x, x_h, N * D);
    castT_f16_kernel<<<blk((long)D * HC), TPB, 0, stream>>>(W1, Wt1, D, HC);
    castT_f16_kernel<<<blk((long)HC * C), TPB, 0, stream>>>(W2, Wt2, HC, C);

    // init accumulators
    fill_kernel<<<blk(N * H), TPB, 0, stream>>>(m1, -1e30f, N * H);
    fill_kernel<<<blk(N * H), TPB, 0, stream>>>(dn1, 0.f, N * H);
    fill_kernel<<<blk((long)N * HC), TPB, 0, stream>>>(out1, 0.f, N * HC);
    fill_kernel<<<blk(N), TPB, 0, stream>>>(m2, -1e30f, N);
    fill_kernel<<<blk(N), TPB, 0, stream>>>(dn2, 0.f, N);
    fill_kernel<<<blk((long)N * C), TPB, 0, stream>>>(out2, 0.f, N * C);
    fill_kernel<<<blk(G * C), TPB, 0, stream>>>(psum, 0.f, G * C);
    fill_kernel<<<blk(G), TPB, 0, stream>>>(pcnt, 0.f, G);

    // ---- layer 1 ----
    {
        dim3 g((N / 16 + WAVES_PER_BLOCK - 1) / WAVES_PER_BLOCK, HC / (NT * 16));
        wmma_gemm_f16_kernel<<<g, 256, 0, stream>>>(x_h, Wt1, h1, N, D, HC);
    }
    alpha_kernel<<<(N * H + 7) / 8, 256, 0, stream>>>(h1, a_src1, a_dst1, as1, ad1, N, H, C);
    edge_logit_max_kernel<<<blk((long)EE * H), TPB, 0, stream>>>(ei, E, N, H, as1, ad1, el1, m1);
    edge_exp_sum_kernel<<<blk((long)EE * H), TPB, 0, stream>>>(ei, E, N, H, m1, el1, dn1);
    aggregate_kernel<<<((long)EE * H + 7) / 8, 256, 0, stream>>>(ei, E, N, H, C, h1, el1, dn1, out1);
    bias_relu_cast_kernel<<<blk((long)N * HC), TPB, 0, stream>>>(out1, b1, h1r, N, HC);

    // ---- layer 2 ----
    {
        dim3 g((N / 16 + WAVES_PER_BLOCK - 1) / WAVES_PER_BLOCK, C / (NT * 16));
        wmma_gemm_f16_kernel<<<g, 256, 0, stream>>>(h1r, Wt2, h2, N, HC, C);
    }
    alpha_kernel<<<(N + 7) / 8, 256, 0, stream>>>(h2, a_src2, a_dst2, as2, ad2, N, 1, C);
    edge_logit_max_kernel<<<blk(EE), TPB, 0, stream>>>(ei, E, N, 1, as2, ad2, el2, m2);
    edge_exp_sum_kernel<<<blk(EE), TPB, 0, stream>>>(ei, E, N, 1, m2, el2, dn2);
    aggregate_kernel<<<(EE + 7) / 8, 256, 0, stream>>>(ei, E, N, 1, C, h2, el2, dn2, out2);

    // ---- pooling + MLP head ----
    pool_accum_kernel<<<blk((long)N * C), TPB, 0, stream>>>(out2, b2, batch, psum, pcnt, N, C);
    final_mlp_kernel<<<G, Hd, 0, stream>>>(psum, pcnt, fc1W, fc1b, fc2W, fc2b,
                                           (float*)d_out, C, Hd);
}
